// Stage_GNN_learner_72035191488519
// MI455X (gfx1250) — compile-verified, matrix operand order checked by reading, not run on hardware
//
#include <hip/hip_runtime.h>
#include <cmath>

// ---------------------------------------------------------------------------
// Types for CDNA5 WMMA (wave32, 16x16x32 bf16 -> f32 accum)
// ---------------------------------------------------------------------------
typedef __attribute__((ext_vector_type(16))) __bf16 v16bf;
typedef __attribute__((ext_vector_type(8)))  __bf16 v8bf;
typedef __attribute__((ext_vector_type(8)))  float  v8f;

__device__ __forceinline__ unsigned short bf16bits(float f) {
  unsigned u = __float_as_uint(f);
  u += 0x7FFFu + ((u >> 16) & 1u);          // round-to-nearest-even
  return (unsigned short)(u >> 16);
}
__device__ __forceinline__ __bf16 f32_to_bf16(float f) {
  union { unsigned short s; __bf16 b; } cv; cv.s = bf16bits(f); return cv.b;
}

// Load one 16x32 operand fragment from an LDS tile row (two ds_load_b128).
// Row layout: 32 K values; lane group hi=0 uses K {0..7,16..23}, hi=1 the rest.
__device__ __forceinline__ v16bf load_frag(const __bf16* rowp, int hi) {
  const v8bf a = *(const v8bf*)(rowp + hi * 8);
  const v8bf b = *(const v8bf*)(rowp + 16 + hi * 8);
  return __builtin_shufflevector(a, b, 0, 1, 2, 3, 4, 5, 6, 7,
                                 8, 9, 10, 11, 12, 13, 14, 15);
}

// ---------------------------------------------------------------------------
// Stage a ROWSx32 operand tile into LDS (row index = M-row for A / N-col for B,
// col index = K). TRANS: logical element (r,k) = src[k*ld + r], else src[r*ld+k].
// Fast path (whole tile in range): float4 global loads, packed LDS stores.
// Slow path: clamp + select, no EXEC divergence.
// ---------------------------------------------------------------------------
template <int ROWS, bool TRANS>
__device__ __forceinline__ void stage_tile(__bf16 (*T)[40],
                                           const float* __restrict__ src,
                                           int ld, int base_r, int Rmax,
                                           int k0, int K, int tid) {
  const bool fast = (base_r + ROWS <= Rmax) && (k0 + 32 <= K);
  if (fast) {
    if (!TRANS) {
      // contiguous along K: one float4 = 4 consecutive k
#pragma unroll
      for (int it = 0; it < (ROWS * 8) / 256; ++it) {
        int v4 = tid + it * 256;            // ROWS*8 vec4s
        int r  = v4 >> 3;
        int kq = (v4 & 7) << 2;
        const float4 f =
            *(const float4*)(src + (size_t)(base_r + r) * ld + (k0 + kq));
        uint2 p;
        p.x = (unsigned)bf16bits(f.x) | ((unsigned)bf16bits(f.y) << 16);
        p.y = (unsigned)bf16bits(f.z) | ((unsigned)bf16bits(f.w) << 16);
        *(uint2*)&T[r][kq] = p;             // 8B LDS store (aligned: pitch 80B)
      }
    } else {
      // contiguous along rows: one float4 = 4 consecutive r
      constexpr int SH = (ROWS == 128) ? 5 : 4;   // log2(ROWS/4)
#pragma unroll
      for (int it = 0; it < (ROWS * 8) / 256; ++it) {
        int v4 = tid + it * 256;
        int kk = v4 >> SH;
        int rq = (v4 & ((ROWS / 4) - 1)) << 2;
        const float4 f =
            *(const float4*)(src + (size_t)(k0 + kk) * ld + (base_r + rq));
        T[rq + 0][kk] = f32_to_bf16(f.x);
        T[rq + 1][kk] = f32_to_bf16(f.y);
        T[rq + 2][kk] = f32_to_bf16(f.z);
        T[rq + 3][kk] = f32_to_bf16(f.w);
      }
    }
  } else {
#pragma unroll
    for (int it = 0; it < (ROWS * 32) / 256; ++it) {
      int e = tid + it * 256;
      int r = e >> 5, kk = e & 31;
      int gr = base_r + r, gk = k0 + kk;
      bool ok = (gr < Rmax) && (gk < K);
      int cr = ok ? gr : 0, ck = ok ? gk : 0;
      size_t idx = TRANS ? ((size_t)ck * ld + cr) : ((size_t)cr * ld + ck);
      float v = src[idx];
      T[r][kk] = f32_to_bf16(ok ? v : 0.f);
    }
  }
}

// ---------------------------------------------------------------------------
// WMMA GEMM: C[MxN] = op(A) @ op(B), fp32 in/out, bf16 operands, f32 accum.
// Block: 256 threads = 8 waves -> 128x64 tile; each wave 2x2 WMMA 16x16 tiles.
// TA: A stored KxM (use A^T). TB: B stored NxK (use B^T).
// ---------------------------------------------------------------------------
template <bool TA, bool TB>
__global__ __launch_bounds__(256) void gemm_wmma(const float* __restrict__ A,
                                                 const float* __restrict__ B,
                                                 float* __restrict__ C,
                                                 int M, int N, int K) {
  __shared__ __bf16 As[128][40];  // M-major, pitch 80B (16B-aligned runs)
  __shared__ __bf16 Bs[64][40];   // N-major (so B frags are contiguous too)

  const int tid  = threadIdx.x;
  const int lane = tid & 31;
  const int wave = tid >> 5;             // 0..7
  const int wr   = (wave >> 1) * 32;     // 0,32,64,96
  const int wc   = (wave & 1) * 32;      // 0,32
  const int bRow = blockIdx.y * 128;
  const int bCol = blockIdx.x * 64;
  const int l    = lane & 15;
  const int hi   = lane >> 4;
  const int lda  = TA ? M : K;
  const int ldb  = TB ? K : N;

  v8f acc[2][2];
#pragma unroll
  for (int i = 0; i < 2; ++i)
#pragma unroll
    for (int j = 0; j < 2; ++j)
#pragma unroll
      for (int v = 0; v < 8; ++v) acc[i][j][v] = 0.f;

  for (int k0 = 0; k0 < K; k0 += 32) {
    stage_tile<128, TA>(As, A, lda, bRow, M, k0, K, tid);
    stage_tile<64, TB>(Bs, B, ldb, bCol, N, k0, K, tid);
    __syncthreads();

    // prefetch next K tile (global_prefetch_b8 on gfx1250)
    if ((k0 + 32) < K && tid < 32) {
      int pr = bRow + tid * 4; if (pr >= M) pr = M - 1;
      size_t aoff = TA ? ((size_t)(k0 + 32) * lda + pr)
                       : ((size_t)pr * lda + (k0 + 32));
      __builtin_prefetch(A + aoff, 0, 1);
      if (tid < 16) {
        int pc = bCol + tid * 4; if (pc >= N) pc = N - 1;
        size_t boff = TB ? ((size_t)pc * ldb + (k0 + 32))
                         : ((size_t)(k0 + 32) * ldb + pc);
        __builtin_prefetch(B + boff, 0, 1);
      }
    }

    v16bf bfrag[2];
#pragma unroll
    for (int tj = 0; tj < 2; ++tj)
      bfrag[tj] = load_frag(&Bs[wc + tj * 16 + l][0], hi);
#pragma unroll
    for (int ti = 0; ti < 2; ++ti) {
      v16bf afrag = load_frag(&As[wr + ti * 16 + l][0], hi);
#pragma unroll
      for (int tj = 0; tj < 2; ++tj) {
        acc[ti][tj] = __builtin_amdgcn_wmma_f32_16x16x32_bf16(
            false, afrag, false, bfrag[tj], (short)0, acc[ti][tj], false, false);
      }
    }
    __syncthreads();
  }

  // store: lanes 0-15 -> rows v (+hi*8), col = lane&15
#pragma unroll
  for (int ti = 0; ti < 2; ++ti)
#pragma unroll
    for (int tj = 0; tj < 2; ++tj)
#pragma unroll
      for (int v = 0; v < 8; ++v) {
        int row = bRow + wr + ti * 16 + v + hi * 8;
        int col = bCol + wc + tj * 16 + l;
        if (row < M && col < N) C[(size_t)row * N + col] = acc[ti][tj][v];
      }
}

// ---------------------------------------------------------------------------
// Elementwise / reduction helper kernels
// ---------------------------------------------------------------------------
#define ACT_NONE 0
#define ACT_RELU 1
#define ACT_SIG  2

__global__ void bias_act_kernel(float* x, const float* b, long long total,
                                int cols, int act) {
  for (long long k = (long long)blockIdx.x * blockDim.x + threadIdx.x; k < total;
       k += (long long)gridDim.x * blockDim.x) {
    float v = x[k] + (b ? b[(int)(k % cols)] : 0.f);
    if (act == ACT_RELU) v = fmaxf(v, 0.f);
    else if (act == ACT_SIG) v = 1.f / (1.f + expf(-v));
    x[k] = v;
  }
}

__global__ void scale_rows_kernel(float* x, const float* d, long long total,
                                  int cols) {
  for (long long k = (long long)blockIdx.x * blockDim.x + threadIdx.x; k < total;
       k += (long long)gridDim.x * blockDim.x)
    x[k] *= d[(int)(k / cols)];
}

__global__ void add_kernel(const float* a, const float* b, float* o,
                           long long total) {
  for (long long k = (long long)blockIdx.x * blockDim.x + threadIdx.x; k < total;
       k += (long long)gridDim.x * blockDim.x)
    o[k] = a[k] + b[k];
}

__global__ void gather_rows_kernel(const float* src, const int* idx, float* out,
                                   long long total, int cols) {
  for (long long k = (long long)blockIdx.x * blockDim.x + threadIdx.x; k < total;
       k += (long long)gridDim.x * blockDim.x) {
    int r = (int)(k / cols), c = (int)(k % cols);
    out[k] = src[(size_t)idx[r] * cols + c];
  }
}

// d = 1/(sum+eos)  (sym=0)  or  1/(sqrt(sum)+eos)  (sym=1)
__global__ void rowsum_inv_kernel(const float* A, int cols, float* d, int sym) {
  __shared__ float red[256];
  int r = blockIdx.x;
  float s = 0.f;
  for (int c = threadIdx.x; c < cols; c += blockDim.x)
    s += A[(size_t)r * cols + c];
  red[threadIdx.x] = s;
  __syncthreads();
  for (int off = 128; off > 0; off >>= 1) {
    if ((int)threadIdx.x < off) red[threadIdx.x] += red[threadIdx.x + off];
    __syncthreads();
  }
  if (threadIdx.x == 0) {
    float t = red[0];
    d[r] = sym ? 1.f / (sqrtf(t) + 1e-10f) : 1.f / (t + 1e-10f);
  }
}

__global__ void l2norm_kernel(const float* x, float* out, int rows, int cols) {
  int r = blockIdx.x * blockDim.x + threadIdx.x;
  if (r >= rows) return;
  float s = 0.f;
  for (int c = 0; c < cols; ++c) { float v = x[(size_t)r * cols + c]; s += v * v; }
  float inv = 1.f / fmaxf(sqrtf(s), 1e-12f);
  for (int c = 0; c < cols; ++c)
    out[(size_t)r * cols + c] = x[(size_t)r * cols + c] * inv;
}

// in-place (a+aT)/2
__global__ void symmetrize_kernel(float* a, int n) {
  int j = blockIdx.x * 16 + threadIdx.x;
  int i = blockIdx.y * 16 + threadIdx.y;
  if (i >= n || j >= n || i > j) return;
  float va = a[(size_t)i * n + j], vb = a[(size_t)j * n + i];
  float m = 0.5f * (va + vb);
  a[(size_t)i * n + j] = m;
  a[(size_t)j * n + i] = m;
}

// a[i][j] *= d[i]*d[j]
__global__ void scale_sym_kernel(float* a, const float* d, int n) {
  int row = blockIdx.y;
  for (int col = blockIdx.x * blockDim.x + threadIdx.x; col < n;
       col += gridDim.x * blockDim.x)
    a[(size_t)row * n + col] *= d[row] * d[col];
}

// per-row: keep top-(Kkeep) via bisection threshold, AND value > eps
__global__ void topk_eps_kernel(float* A, int n, int Kkeep, float eps) {
  __shared__ int   red[256];
  __shared__ float bounds[2];
  int r = blockIdx.x;
  float* row = A + (size_t)r * n;
  if (threadIdx.x == 0) { bounds[0] = -2.f; bounds[1] = 2.f; }
  __syncthreads();
  for (int it = 0; it < 40; ++it) {
    float mid = 0.5f * (bounds[0] + bounds[1]);
    int c = 0;
    for (int j = threadIdx.x; j < n; j += blockDim.x) c += (row[j] > mid) ? 1 : 0;
    red[threadIdx.x] = c;
    __syncthreads();
    for (int off = 128; off > 0; off >>= 1) {
      if ((int)threadIdx.x < off) red[threadIdx.x] += red[threadIdx.x + off];
      __syncthreads();
    }
    if (threadIdx.x == 0) { if (red[0] >= Kkeep) bounds[0] = mid; else bounds[1] = mid; }
    __syncthreads();
  }
  float th = bounds[0];
  for (int j = threadIdx.x; j < n; j += blockDim.x) {
    float v = row[j];
    row[j] = (v > th && v > eps) ? v : 0.f;
  }
}

// exact rank selection (strict deterministic tie-break): keep rank < kk
__global__ void rank_select_kernel(const float* y, int n, int kk, int* keep) {
  int i = blockIdx.x * blockDim.x + threadIdx.x;
  if (i >= n) return;
  float yi = y[i];
  int c = 0;
  for (int j = 0; j < n; ++j) {
    float yj = y[j];
    c += ((yj > yi) || (yj == yi && j < i)) ? 1 : 0;
  }
  keep[i] = (c < kk) ? 1 : 0;
}

// serial compaction (ascending index order == reference argsort(idx))
__global__ void compact_kernel(const int* keep, int n, const int* oldmap,
                               int* out) {
  if (threadIdx.x == 0 && blockIdx.x == 0) {
    int p = 0;
    for (int i = 0; i < n; ++i)
      if (keep[i]) out[p++] = oldmap ? oldmap[i] : i;
  }
}

__global__ void seti_kernel(int* x, int n, int val) {
  int i = blockIdx.x * blockDim.x + threadIdx.x;
  if (i < n) x[i] = val;
}
__global__ void scatteri_kernel(int* x, const int* idx, int m, int val) {
  int i = blockIdx.x * blockDim.x + threadIdx.x;
  if (i < m) x[idx[i]] = val;
}

// concat [e | v_pos_emb[vpos]] -> [rows x (c1+c2)]
__global__ void concat_kernel(const float* e, const int* vpos, const float* vpe,
                              float* out, long long total, int c1, int c2) {
  int cols = c1 + c2;
  for (long long k = (long long)blockIdx.x * blockDim.x + threadIdx.x; k < total;
       k += (long long)gridDim.x * blockDim.x) {
    int r = (int)(k / cols), c = (int)(k % cols);
    out[k] = (c < c1) ? e[(size_t)r * c1 + c]
                      : vpe[(size_t)vpos[r] * c2 + (c - c1)];
  }
}

// adj_full[sub] = (learned[sub]*r + adj_full[sub]*(1-r)).T  -- paired in-place
__global__ void modify_sub_kernel(const float* learned, float* adjf,
                                  const int* idx, int m, int N, float r) {
  int b = blockIdx.x * 16 + threadIdx.x;
  int a = blockIdx.y * 16 + threadIdx.y;
  if (a >= m || b >= m || a > b) return;
  int ia = idx[a], ib = idx[b];
  float o1 = adjf[(size_t)ia * N + ib], o2 = adjf[(size_t)ib * N + ia];
  float l1 = learned[(size_t)ia * N + ib], l2 = learned[(size_t)ib * N + ia];
  adjf[(size_t)ia * N + ib] = l2 * r + o2 * (1.f - r);
  adjf[(size_t)ib * N + ia] = l1 * r + o1 * (1.f - r);
}

// pred (in place over adj_full slot) = thresh(0.8*learned + 0.2*adj_full)
__global__ void fuse_kernel(const float* learned, float* predf, long long total) {
  for (long long k = (long long)blockIdx.x * blockDim.x + threadIdx.x; k < total;
       k += (long long)gridDim.x * blockDim.x) {
    float v = 0.8f * learned[k] + 0.2f * predf[k];
    predf[k] = (v < 0.002f) ? 0.f : v;
  }
}

// ---------------------------------------------------------------------------
// Host orchestration
// ---------------------------------------------------------------------------
static inline dim3 gs1d(long long total) {
  long long b = (total + 255) / 256;
  if (b > 8192) b = 8192;
  if (b < 1) b = 1;
  return dim3((unsigned)b);
}

static void gemm(int TA, int TB, const float* A, const float* B, float* C,
                 int M, int N, int K, hipStream_t s) {
  dim3 g((N + 63) / 64, (M + 127) / 128), blk(256);
  if (!TA && !TB)      gemm_wmma<false, false><<<g, blk, 0, s>>>(A, B, C, M, N, K);
  else if (TA && !TB)  gemm_wmma<true,  false><<<g, blk, 0, s>>>(A, B, C, M, N, K);
  else if (!TA && TB)  gemm_wmma<false, true ><<<g, blk, 0, s>>>(A, B, C, M, N, K);
  else                 gemm_wmma<true,  true ><<<g, blk, 0, s>>>(A, B, C, M, N, K);
}

extern "C" void kernel_launch(void* const* d_in, const int* in_sizes, int n_in,
                              void* d_out, int out_size, void* d_ws,
                              size_t ws_size, hipStream_t stream) {
  (void)in_sizes; (void)out_size; (void)ws_size;
  if (n_in < 27) return;

  const int n = 4096, ISZ = 256, OSZ = 64, VD = 16;
  const long long NN = (long long)n * n;

  // ---- inputs (setup_inputs dict order, params flattened depth-first) ----
  const float* features = (const float*)d_in[0];
  const float* adj      = (const float*)d_in[1];
  const float* i0w1 = (const float*)d_in[2];  const float* i0b1 = (const float*)d_in[3];
  const float* i0w2 = (const float*)d_in[4];  const float* i0b2 = (const float*)d_in[5];
  const float* i1w1 = (const float*)d_in[6];  const float* i1b1 = (const float*)d_in[7];
  const float* i1w2 = (const float*)d_in[8];  const float* i1b2 = (const float*)d_in[9];
  const float* scw1 = (const float*)d_in[10]; const float* scb1 = (const float*)d_in[11];
  const float* scw2 = (const float*)d_in[12]; const float* scb2 = (const float*)d_in[13];
  const float* u0w1 = (const float*)d_in[14]; const float* u0b1 = (const float*)d_in[15];
  const float* u0w2 = (const float*)d_in[16]; const float* u0b2 = (const float*)d_in[17];
  const float* u1w1 = (const float*)d_in[18]; const float* u1b1 = (const float*)d_in[19];
  const float* u1w2 = (const float*)d_in[20]; const float* u1b2 = (const float*)d_in[21];
  const float* vpe  = (const float*)d_in[22];
  const float* m1w  = (const float*)d_in[23]; const float* m1b  = (const float*)d_in[24];
  const float* m2w  = (const float*)d_in[25]; const float* m2b  = (const float*)d_in[26];

  // ---- outputs (learned, prediction, adj_cur, new_features) ----
  float* LEARNED = (float*)d_out;
  float* PRED    = LEARNED + NN;       // doubles as evolving adj_full
  float* ADJCUR  = PRED + NN;          // 1024 x 4096
  float* NEWF    = ADJCUR + (long long)1024 * n;  // 1024 x 256

  // ---- workspace layout (floats) ----
  float* W = (float*)d_ws;
  size_t off = 0;
  float* EMB    = W + off; off += (size_t)n * OSZ;  // embeddings == cur == down_outs
  float* EOUT   = W + off; off += (size_t)n * OSZ;
  float* EN     = W + off; off += (size_t)n * OSZ;
  float* H      = W + off; off += (size_t)n * OSZ;
  float* T1     = W + off; off += (size_t)n * 256;
  float* T2     = W + off; off += (size_t)n * 256;
  float* T3     = W + off; off += (size_t)n * 256;
  float* CONCAT = W + off; off += (size_t)n * (OSZ + VD);
  float* D      = W + off; off += (size_t)n;
  float* ADJSEL = W + off; off += (size_t)2048 * n;
  int* VPOS = (int*)(W + off); off += (size_t)n;
  int* KEEP = (int*)(W + off); off += (size_t)n;
  int* IDX0 = (int*)(W + off); off += 2048;
  int* IDX1 = (int*)(W + off); off += 1024;

  dim3 b256(256);
  auto bias_act = [&](float* x, const float* bb, long long rows, int cols, int act) {
    bias_act_kernel<<<gs1d(rows * cols), b256, 0, stream>>>(x, bb, rows * cols, cols, act);
  };
  auto scale_rows = [&](float* x, long long rows, int cols) {
    scale_rows_kernel<<<gs1d(rows * cols), b256, 0, stream>>>(x, D, rows * cols, cols);
  };

  // ======== internal[0]: h = relu(hgnn_conv(features, adj)) ========
  gemm(0, 0, features, i0w1, T1, n, 32, ISZ, stream);
  bias_act(T1, i0b1, n, 32, ACT_RELU);
  gemm(1, 0, adj, T1, T2, n, 32, n, stream);   // adj.T @ h1
  gemm(0, 0, adj, T2, T1, n, 32, n, stream);   // adj @ ..
  gemm(0, 0, T1, i0w2, H, n, OSZ, 32, stream);
  bias_act(H, i0b2, n, OSZ, ACT_RELU);

  // ======== internal[1]: embeddings ========
  gemm(0, 0, H, i1w1, T1, n, 32, OSZ, stream);
  bias_act(T1, i1b1, n, 32, ACT_RELU);
  gemm(1, 0, adj, T1, T2, n, 32, n, stream);
  gemm(0, 0, adj, T2, T1, n, 32, n, stream);
  gemm(0, 0, T1, i1w2, EMB, n, OSZ, 32, stream);
  bias_act(EMB, i1b2, n, OSZ, ACT_NONE);

  // ======== pool stage 0: score on EMB with G = adj ========
  gemm(0, 0, EMB, scw1, T1, n, 32, OSZ, stream);
  bias_act(T1, scb1, n, 32, ACT_RELU);
  gemm(1, 0, adj, T1, T2, n, 32, n, stream);
  gemm(0, 0, adj, T2, T1, n, 32, n, stream);
  gemm(0, 0, T1, scw2, T2, n, 1, 32, stream);
  bias_act(T2, scb2, n, 1, ACT_RELU);          // y0 in T2[0..n)
  rank_select_kernel<<<gs1d(n), b256, 0, stream>>>(T2, n, 2048, KEEP);
  compact_kernel<<<1, 1, 0, stream>>>(KEEP, n, (const int*)nullptr, IDX0);
  gather_rows_kernel<<<gs1d((long long)2048 * n), b256, 0, stream>>>(
      adj, IDX0, ADJSEL, (long long)2048 * n, n);

  // ======== pool stage 1: score on EMB[IDX0] with row-normalized ADJSEL ====
  rowsum_inv_kernel<<<2048, 256, 0, stream>>>(ADJSEL, n, D, 0);  // row mode
  gather_rows_kernel<<<gs1d((long long)2048 * OSZ), b256, 0, stream>>>(
      EMB, IDX0, T3, (long long)2048 * OSZ, OSZ);
  gemm(0, 0, T3, scw1, T1, 2048, 32, OSZ, stream);
  bias_act(T1, scb1, 2048, 32, ACT_RELU);
  scale_rows(T1, 2048, 32);                    // An.T @ s = a.T @ (d*s)
  gemm(1, 0, ADJSEL, T1, T2, n, 32, 2048, stream);
  gemm(0, 0, ADJSEL, T2, T1, 2048, 32, n, stream);
  scale_rows(T1, 2048, 32);                    // An @ s = d * (a@s)
  gemm(0, 0, T1, scw2, T2, 2048, 1, 32, stream);
  bias_act(T2, scb2, 2048, 1, ACT_RELU);
  rank_select_kernel<<<gs1d(2048), b256, 0, stream>>>(T2, 2048, 1024, KEEP);
  compact_kernel<<<1, 1, 0, stream>>>(KEEP, 2048, IDX0, IDX1);

  // ======== vpos + concat (EMB fixed -> CONCAT reused for all apply_vpos) ==
  seti_kernel<<<gs1d(n), b256, 0, stream>>>(VPOS, n, 0);
  scatteri_kernel<<<gs1d(2048), b256, 0, stream>>>(VPOS, IDX0, 2048, 1);
  scatteri_kernel<<<gs1d(1024), b256, 0, stream>>>(VPOS, IDX1, 1024, 2);
  concat_kernel<<<gs1d((long long)n * (OSZ + VD)), b256, 0, stream>>>(
      EMB, VPOS, vpe, CONCAT, (long long)n * (OSZ + VD), OSZ, VD);

  auto apply_vpos = [&](float* out) {
    gemm(0, 0, CONCAT, m1w, T1, n, OSZ, OSZ + VD, stream);
    bias_act(T1, m1b, n, OSZ, ACT_SIG);
    gemm(0, 0, T1, m2w, out, n, OSZ, OSZ, stream);
    bias_act(out, m2b, n, OSZ, ACT_NONE);
  };
  auto learned_from = [&](float* e) {   // e += cur; l2norm; e@e.T; topk+eps
    add_kernel<<<gs1d((long long)n * OSZ), b256, 0, stream>>>(
        e, EMB, e, (long long)n * OSZ);
    l2norm_kernel<<<gs1d(n), b256, 0, stream>>>(e, EN, n, OSZ);
    gemm(0, 1, EN, EN, LEARNED, n, n, OSZ, stream);
    topk_eps_kernel<<<n, 256, 0, stream>>>(LEARNED, n, 31, 0.1f);
  };
  auto sym_norm_learned = [&]() {
    symmetrize_kernel<<<dim3(n / 16, n / 16), dim3(16, 16), 0, stream>>>(LEARNED, n);
    rowsum_inv_kernel<<<n, 256, 0, stream>>>(LEARNED, n, D, 1);
    scale_sym_kernel<<<dim3(16, n), b256, 0, stream>>>(LEARNED, D, n);
  };
  // up conv with implicit A = D*adj_full*D (D precomputed in `D`)
  auto up_conv = [&](const float* ein, const float* w1, const float* b1,
                     const float* w2, const float* b2, float* out, int act) {
    gemm(0, 0, ein, w1, T1, n, 256, OSZ, stream);
    bias_act(T1, b1, n, 256, ACT_RELU);
    scale_rows(T1, n, 256);                    // A.T@x = D a.T (D x)
    gemm(1, 0, PRED, T1, T2, n, 256, n, stream);
    scale_rows(T2, n, 256);                    // finish A.T@x
    scale_rows(T2, n, 256);                    // pre-scale for A@y
    gemm(0, 0, PRED, T2, T1, n, 256, n, stream);
    scale_rows(T1, n, 256);                    // finish A@y
    gemm(0, 0, T1, w2, out, n, OSZ, 256, stream);
    bias_act(out, b2, n, OSZ, act);
  };

  // ======== emb = apply_vpos(emb) + cur ; first learned_adj ========
  apply_vpos(EOUT);
  learned_from(EOUT);

  // adj_full := adj  (lives in PRED output slot)
  hipMemcpyAsync(PRED, adj, NN * sizeof(float), hipMemcpyDeviceToDevice, stream);

  // ======== up loop: j = 1 (IDX1, m=1024), then j = 0 (IDX0, m=2048) ======
  const int* idxs[2] = {IDX1, IDX0};
  const int  ms[2]   = {1024, 2048};
  for (int t = 0; t < 2; ++t) {
    sym_norm_learned();
    modify_sub_kernel<<<dim3((ms[t] + 15) / 16, (ms[t] + 15) / 16),
                        dim3(16, 16), 0, stream>>>(LEARNED, PRED, idxs[t],
                                                   ms[t], n, 0.8f);
    rowsum_inv_kernel<<<n, 256, 0, stream>>>(PRED, n, D, 1);  // D for A=norm(adj_full)
    apply_vpos(EOUT);                                          // e = apply_vpos(down_out)
    up_conv(EOUT, u0w1, u0b1, u0w2, u0b2, T3, ACT_RELU);
    up_conv(T3,   u1w1, u1b1, u1w2, u1b2, EOUT, ACT_NONE);
    learned_from(EOUT);
  }

  // ======== final normalize + fusion + gathers ========
  sym_norm_learned();
  fuse_kernel<<<gs1d(NN), b256, 0, stream>>>(LEARNED, PRED, NN);
  gather_rows_kernel<<<gs1d((long long)1024 * n), b256, 0, stream>>>(
      adj, IDX1, ADJCUR, (long long)1024 * n, n);
  gather_rows_kernel<<<gs1d((long long)1024 * ISZ), b256, 0, stream>>>(
      features, IDX1, NEWF, (long long)1024 * ISZ, ISZ);
}
